// BNLSTM_84834194031071
// MI455X (gfx1250) — matrix operand order repaired and checked by reading, not compiled, not run on hardware
//
#include <hip/hip_runtime.h>

// ---------------------------------------------------------------------------
// BN-LSTM for MI455X (gfx1250).
//  - bf16 WMMA (v_wmma_f32_16x16x32_bf16) for both projections
//  - async global->LDS staging of B tiles (ASYNCcnt, s_wait_asynccnt),
//    double-buffered, shared by all 4 waves of a block
//  - LDS tile transpose => coalesced transposed C stores, coalesced BN reads
//  - ih projection precomputed CHUNK timesteps at a time (M up to 4096)
// B=64, S=512, D=512, H=1024, 4H=4096.
// ---------------------------------------------------------------------------

typedef __attribute__((ext_vector_type(16))) __bf16 v16bf;
typedef __attribute__((ext_vector_type(8)))  float  v8f;

union BfFrag { v16bf v; unsigned int u[8]; };

#define BATCH 64
#define SEQ   512
#define DIN   512
#define HID   1024
#define G4    4096   // 4*HID
#define KC    32     // K per WMMA chunk

__device__ __forceinline__ unsigned short f32_to_bf16(float f) {
    union { float f; unsigned int u; } x; x.f = f;
    unsigned int r = x.u + 0x7FFFu + ((x.u >> 16) & 1u);   // round-to-nearest-even
    return (unsigned short)(r >> 16);
}

__device__ __forceinline__ float sigf(float x) { return 1.0f / (1.0f + __expf(-x)); }

// -------------------------- conversion kernels -----------------------------

__global__ void cvt_bf16_vec4(const float* __restrict__ in,
                              unsigned short* __restrict__ out, int n) {
    int i = (blockIdx.x * blockDim.x + threadIdx.x) * 4;
    if (i + 3 < n) {
        float4 f = *(const float4*)(in + i);
        unsigned short s[4];
        s[0] = f32_to_bf16(f.x); s[1] = f32_to_bf16(f.y);
        s[2] = f32_to_bf16(f.z); s[3] = f32_to_bf16(f.w);
        *(uint2*)(out + i) = *(const uint2*)s;
    }
}

// fp32 [rows, cols] row-major -> bf16 [cols, rows] (B stored K-contiguous)
__global__ void cvt_bf16_transpose(const float* __restrict__ in,
                                   unsigned short* __restrict__ out,
                                   int rows, int cols) {
    int c = blockIdx.x * blockDim.x + threadIdx.x;
    int r = blockIdx.y;
    if (c < cols) out[(size_t)c * rows + r] = f32_to_bf16(in[(size_t)r * cols + c]);
}

// c state stored transposed [HID][64]; h state kept as bf16 [64][HID] (GEMM A)
__global__ void init_state(const float* __restrict__ h0, const float* __restrict__ c0,
                           float* __restrict__ cT, unsigned short* __restrict__ hbf) {
    int idx = blockIdx.x * blockDim.x + threadIdx.x;   // 0 .. 64*1024-1
    int j = idx >> 6, b = idx & 63;
    cT[idx] = c0[j];
    hbf[(size_t)b * HID + j] = f32_to_bf16(h0[j]);
}

// ----------------------------- WMMA GEMM -----------------------------------
// Ct[N, M] (transposed, fp32) = A[M, K](bf16) * B (bf16 as BT[N, K]).
// Block = 128 threads = 4 waves covering a 64x64 tile of C; one wave = 16x64.
// B tiles (64 N x 32 K) are async-staged into LDS, double-buffered, and shared
// by all 4 waves. Output goes through an LDS transpose for coalesced stores.
// rowmap=1: A row for logical m is ((m&63)*SEQ + t0 + (m>>6)) of x (chunk GEMM)
__global__ void __launch_bounds__(128)
wmma_gemm_bt(const unsigned short* __restrict__ A, int rowmap, int t0, int lda,
             const unsigned short* __restrict__ BT, int K,
             float* __restrict__ Ct, int M) {
    __shared__ unsigned short sB[2][64][KC];   // 2 x 4 KB
    __shared__ float sC[64][65];               // padded: conflict-free transpose

    const int tid  = threadIdx.x;
    const int lane = tid & 31;
    const int wv   = tid >> 5;                 // 0..3: local M-tile
    const int half = lane >> 4;
    const int ml   = lane & 15;
    const int nblk = blockIdx.x * 64;
    const int mblk = blockIdx.y * 64;

    const int m = mblk + wv * 16 + ml;
    const size_t arow = rowmap ? ((size_t)(m & 63) * SEQ + t0 + (m >> 6)) * DIN
                               : (size_t)m * lda;

    // async-stage one 64x32 bf16 B tile into LDS: 2 x b128 per thread
    auto stage = [&](int buf, int k0) {
#pragma unroll
        for (int i = 0; i < 2; ++i) {
            int seg  = tid * 2 + i;            // 0..255 16-byte segments
            int n    = seg >> 2;
            int part = (seg & 3) * 8;          // offset in shorts
            unsigned gofs = (unsigned)((((nblk + n) * K) + k0 + part) * 2);
            unsigned lofs = (unsigned)(uintptr_t)&sB[buf][n][part];
            asm volatile("global_load_async_to_lds_b128 %0, %1, %2"
                         :: "v"(lofs), "v"(gofs), "s"(BT) : "memory");
        }
    };

    v8f acc[4] = {};
    stage(0, 0);
    for (int k0 = 0; k0 < K; k0 += KC) {
        const int buf = (k0 >> 5) & 1;
        if (k0 + KC < K) {
            stage(buf ^ 1, k0 + KC);                       // prefetch next tile
            __builtin_prefetch(A + arow + k0 + KC, 0, 1);  // global_prefetch_b8
            asm volatile("s_wait_asynccnt 0x2" ::: "memory");  // current tile done
        } else {
            asm volatile("s_wait_asynccnt 0x0" ::: "memory");
        }
        __syncthreads();

        // A fragment 16x32 bf16: lanes 0-15 K 0-15; lanes 16-31 K 16-31.
        BfFrag a;
#pragma unroll
        for (int v = 0; v < 8; ++v) {
            int k = k0 + 2 * v + 16 * half;
            a.u[v] = *(const unsigned int*)(A + arow + k);
        }
#pragma unroll
        for (int nt = 0; nt < 4; ++nt) {
            // B fragment from LDS: VGPR v holds K = 2v + 16*half, contiguous.
            BfFrag b;
            const unsigned short* bp = &sB[buf][nt * 16 + ml][16 * half];
#pragma unroll
            for (int v = 0; v < 8; ++v)
                b.u[v] = *(const unsigned int*)(bp + 2 * v);
            acc[nt] = __builtin_amdgcn_wmma_f32_16x16x32_bf16(
                false, a.v, false, b.v, (short)0, acc[nt], false, false);
        }
        __syncthreads();
    }

    // C/D layout: VGPR r holds M = r + 8*half, N = ml. Transpose via LDS.
#pragma unroll
    for (int nt = 0; nt < 4; ++nt)
#pragma unroll
        for (int r = 0; r < 8; ++r)
            sC[wv * 16 + r + 8 * half][nt * 16 + ml] = acc[nt][r];
    __syncthreads();
    for (int e = tid; e < 64 * 64; e += 128) {
        int ml2 = e & 63;                      // consecutive tid -> consecutive m
        int nl2 = e >> 6;
        Ct[(size_t)(nblk + nl2) * M + mblk + ml2] = sC[ml2][nl2];
    }
}

// --------------------------- BN + gate step --------------------------------

__device__ __forceinline__ float wave_sum(float v) {
#pragma unroll
    for (int off = 16; off > 0; off >>= 1) v += __shfl_xor(v, off, 32);
    return v;
}

// BatchNorm over 64 batch values held as 2 per lane across a wave32.
__device__ __forceinline__ void bn2(float v0, float v1, float& o0, float& o1) {
    float mean = wave_sum(v0 + v1) * (1.0f / 64.0f);
    float d0 = v0 - mean, d1 = v1 - mean;
    float var = wave_sum(d0 * d0 + d1 * d1) * (1.0f / 64.0f);
    float inv = rsqrtf(var + 1e-5f);
    o0 = d0 * inv; o1 = d1 * inv;
}

// One wave per hidden column j; lane holds batch rows (lane, lane+32).
// ihT: [4096][CHUNK*64] (m = tt*64 + b), hhT: [4096][64], cT: [HID][64].
__global__ void __launch_bounds__(256)
bn_lstm_step(const float* __restrict__ ihT, int Mih, int tt,
             const float* __restrict__ hhT,
             const float* __restrict__ bias,
             const float* __restrict__ gamma, const float* __restrict__ beta,
             float* __restrict__ cT, unsigned short* __restrict__ hbf,
             float* __restrict__ hid_t,     // &hiddens[0, t, 0]
             float* __restrict__ hfin, float* __restrict__ cfin, int last) {
    const int lane = threadIdx.x & 31;
    const int j = blockIdx.x * (blockDim.x >> 5) + (threadIdx.x >> 5);
    const int r0 = lane, r1 = lane + 32;

    float g[4][2];
#pragma unroll
    for (int q = 0; q < 4; ++q) {            // gate order: i, f, o, g
        int col = q * HID + j;
        const float* ip = ihT + (size_t)col * Mih + (size_t)tt * 64;  // coalesced
        const float* hp = hhT + (size_t)col * 64;                     // coalesced
        float na0, na1, nb0, nb1;
        bn2(ip[r0], ip[r1], na0, na1);
        bn2(hp[r0], hp[r1], nb0, nb1);
        float bs = bias[col];
        g[q][0] = na0 + nb0 + bs;
        g[q][1] = na1 + nb1 + bs;
    }

    float* cp = cT + (size_t)j * 64;
    float nc0 = sigf(g[1][0]) * cp[r0] + sigf(g[0][0]) * tanhf(g[3][0]);
    float nc1 = sigf(g[1][1]) * cp[r1] + sigf(g[0][1]) * tanhf(g[3][1]);

    float bn0, bn1;
    bn2(nc0, nc1, bn0, bn1);
    float gm = gamma[j], bt = beta[j];
    float nh0 = sigf(g[2][0]) * tanhf(gm * bn0 + bt);
    float nh1 = sigf(g[2][1]) * tanhf(gm * bn1 + bt);

    cp[r0] = nc0;  cp[r1] = nc1;
    hbf[(size_t)r0 * HID + j] = f32_to_bf16(nh0);
    hbf[(size_t)r1 * HID + j] = f32_to_bf16(nh1);

    const size_t bstride = (size_t)SEQ * HID;          // output layout [B,S,H]
    hid_t[(size_t)r0 * bstride + j] = nh0;
    hid_t[(size_t)r1 * bstride + j] = nh1;

    if (last) {
        hfin[(size_t)r0 * HID + j] = nh0;  hfin[(size_t)r1 * HID + j] = nh1;
        cfin[(size_t)r0 * HID + j] = nc0;  cfin[(size_t)r1 * HID + j] = nc1;
    }
}

// ------------------------------- launcher ----------------------------------

extern "C" void kernel_launch(void* const* d_in, const int* in_sizes, int n_in,
                              void* d_out, int out_size, void* d_ws, size_t ws_size,
                              hipStream_t stream) {
    const float* x     = (const float*)d_in[0];   // [64, 512, 512]
    const float* w_ih  = (const float*)d_in[1];   // [512, 4096]
    const float* w_hh  = (const float*)d_in[2];   // [1024, 4096]
    const float* bias  = (const float*)d_in[3];   // [4096]
    const float* gamma = (const float*)d_in[4];   // [1024]
    const float* beta  = (const float*)d_in[5];   // [1024]
    const float* h0    = (const float*)d_in[6];   // [1024]
    const float* c0    = (const float*)d_in[7];   // [1024]

    float* out     = (float*)d_out;
    float* hiddens = out;                                     // [64, 512, 1024]
    float* hfin    = out + (size_t)BATCH * SEQ * HID;
    float* cfin    = hfin + (size_t)BATCH * HID;

    char* ws = (char*)d_ws;
    unsigned short* x_bf = (unsigned short*)ws; ws += (size_t)BATCH * SEQ * DIN * 2; // 32 MB
    unsigned short* wihT = (unsigned short*)ws; ws += (size_t)G4 * DIN * 2;          //  4 MB
    unsigned short* whhT = (unsigned short*)ws; ws += (size_t)G4 * HID * 2;          //  8 MB
    float* hhT = (float*)ws;          ws += (size_t)G4 * BATCH * 4;                  //  1 MB
    float* cT  = (float*)ws;          ws += (size_t)HID * BATCH * 4;
    unsigned short* h_bf = (unsigned short*)ws; ws += (size_t)BATCH * HID * 2;
    size_t fixed = (size_t)(ws - (char*)d_ws);
    float* ihT = (float*)ws;                       // chunk buffer, sized below

    // Largest power-of-two chunk of timesteps whose ih results fit in ws.
    size_t avail = (ws_size > fixed) ? (ws_size - fixed) : 0;
    int chunk = 64;
    while (chunk > 1 && (size_t)G4 * chunk * 64 * 4 > avail) chunk >>= 1;

    {
        int n = BATCH * SEQ * DIN;                    // 16.7M
        cvt_bf16_vec4<<<n / (256 * 4), 256, 0, stream>>>(x, x_bf, n);
        cvt_bf16_transpose<<<dim3(G4 / 256, DIN), 256, 0, stream>>>(w_ih, wihT, DIN, G4);
        cvt_bf16_transpose<<<dim3(G4 / 256, HID), 256, 0, stream>>>(w_hh, whhT, HID, G4);
        init_state<<<(BATCH * HID) / 256, 256, 0, stream>>>(h0, c0, cT, h_bf);
    }

    for (int t0 = 0; t0 < SEQ; t0 += chunk) {
        // ih for `chunk` timesteps in one launch: M = chunk*64 rows.
        wmma_gemm_bt<<<dim3(G4 / 64, chunk), 128, 0, stream>>>(
            x_bf, /*rowmap=*/1, t0, 0, wihT, DIN, ihT, chunk * 64);
        for (int tt = 0; tt < chunk; ++tt) {
            int t = t0 + tt;
            wmma_gemm_bt<<<dim3(G4 / 64, 1), 128, 0, stream>>>(
                h_bf, /*rowmap=*/0, 0, HID, whhT, HID, hhT, 64);
            bn_lstm_step<<<HID / 8, 256, 0, stream>>>(
                ihT, chunk * 64, tt, hhT, bias, gamma, beta, cT, h_bf,
                hiddens + (size_t)t * HID, hfin, cfin, (t == SEQ - 1) ? 1 : 0);
        }
    }
}
